// MultiHeadAttention_9569187135619
// MI455X (gfx1250) — matrix operand order repaired and checked by reading, not compiled, not run on hardware
//
#include <hip/hip_runtime.h>
#include <hip/hip_bf16.h>

// MHA forward, S=2048 B=4 E=1024 H=16 D=64. Compute-bound (~137 GFLOP vs ~150MB
// traffic at 23.3 TB/s) -> all matmuls on v_wmma_f32_16x16x32_bf16, f32 accum.
// f32->bf16 conversion fused into LDS staging. Attention tiles staged with
// gfx1250 async-to-LDS copies (ASYNCcnt), GEMM K-loop uses global_prefetch.

typedef __bf16 v16bf __attribute__((ext_vector_type(16)));
typedef float  v8f   __attribute__((ext_vector_type(8)));

#define SEQ   2048
#define BATCH 4
#define EMB   1024
#define HEADS 16
#define HD    64

__device__ __forceinline__ unsigned short f2bf(float x) {
  unsigned u = __float_as_uint(x);
  u += 0x7FFFu + ((u >> 16) & 1u);           // round-to-nearest-even
  return (unsigned short)(u >> 16);
}
__device__ __forceinline__ unsigned pack2bf(float a, float b) {
  return (unsigned)f2bf(a) | ((unsigned)f2bf(b) << 16);
}

// Async 16B copy global -> LDS (per lane). ASYNCcnt-tracked, no VGPR data path.
// LDS aperture: flat addr[31:0] == LDS byte offset, so truncate the pointer.
__device__ __forceinline__ void async_lds_b128(unsigned short* dstLds,
                                               const unsigned short* srcGlobal) {
  unsigned ldsOff = (unsigned)(uintptr_t)dstLds;
  unsigned long long ga = (unsigned long long)(uintptr_t)srcGlobal;
  asm volatile("global_load_async_to_lds_b128 %0, %1, off"
               :: "v"(ldsOff), "v"(ga) : "memory");
}
__device__ __forceinline__ void wait_async_all() {
  asm volatile("s_wait_asynccnt 0x0" ::: "memory");
}

union FragU { v16bf v; unsigned u[8]; unsigned short s[16]; };

// A fragment (16x32 bf16), smem layout [m][k], stride in elements.
// k order used here (contiguous, same as B): lane = M (mod 16); lanes 0-15 K 0..15, lanes 16-31 K 16..31.
__device__ __forceinline__ v16bf load_a_frag(const unsigned short* smem, int stride,
                                             int m_base, int k_base, int lane) {
  FragU f;
  const unsigned short* p =
      smem + (m_base + (lane & 15)) * stride + k_base + ((lane >> 4) << 4);
#pragma unroll
  for (int i = 0; i < 8; ++i) f.u[i] = *(const unsigned*)(p + 2 * i);
  return f.v;
}

// B fragment (32x16 bf16), B[k][n] stored n-major: smem[n*stride + k].
// k order used here (contiguous, same as A): lane = N (mod 16); lanes 0-15 K 0..15, lanes 16-31 K 16..31.
__device__ __forceinline__ v16bf load_b_frag(const unsigned short* smem, int stride,
                                             int n_base, int k_base, int lane) {
  FragU f;
  const unsigned short* p =
      smem + (n_base + (lane & 15)) * stride + k_base + ((lane >> 4) << 4);
#pragma unroll
  for (int i = 0; i < 8; ++i) f.u[i] = *(const unsigned*)(p + 2 * i);
  return f.v;
}

// B fragment from k-major storage: smem[k*stride + n]  (used for V tiles).
__device__ __forceinline__ v16bf load_b_frag_T(const unsigned short* smem, int stride,
                                               int k_base, int n_base, int lane) {
  FragU f;
  const int n  = n_base + (lane & 15);
  const int k0 = k_base + ((lane >> 4) << 4);
#pragma unroll
  for (int i = 0; i < 8; ++i) {
    f.s[2 * i]     = smem[(k0 + 2 * i) * stride + n];
    f.s[2 * i + 1] = smem[(k0 + 2 * i + 1) * stride + n];
  }
  return f.v;
}

// ---------------------------------------------------------------------------
// GEMM: Y = X @ W^T.  X:[M,K] (f32 or bf16), W:[N,K] f32 row-major.
// MODE 0: X f32, write bf16 head-major [B,H,S,D] (projection kernels).
// MODE 1: X bf16, write f32 row-major [M,N]      (output projection).
// Block: 128x128 tile, BK=32, 256 threads = 8 waves, wave -> 32(M)x64(N).
// ---------------------------------------------------------------------------
template <int MODE>
__global__ __launch_bounds__(256) void gemm_xwt_kernel(const void* __restrict__ Aptr,
                                                       const float* __restrict__ W,
                                                       void* __restrict__ Out) {
  constexpr int N = EMB, K = EMB;
  constexpr int BK = 32, AST = BK + 8;           // padded LDS stride
  __shared__ unsigned short Xs[128 * AST];
  __shared__ unsigned short Ws[128 * AST];
  const int t = threadIdx.x, lane = t & 31, w = t >> 5;
  const int wm = (w & 3) << 5, wn = (w >> 2) << 6;
  const int m0 = blockIdx.x << 7, n0 = blockIdx.y << 7;

  v8f acc[2][4];
#pragma unroll
  for (int i = 0; i < 2; ++i)
#pragma unroll
    for (int j = 0; j < 4; ++j) acc[i][j] = {};

  for (int k0 = 0; k0 < K; k0 += BK) {
    if (MODE == 0) {                              // f32 A tile -> bf16 LDS
      const float* A = (const float*)Aptr;
#pragma unroll
      for (int i = 0; i < 4; ++i) {
        int q = t + (i << 8);
        int row = q >> 3, c = (q & 7) << 2;
        float4 x = *(const float4*)(A + (size_t)(m0 + row) * K + k0 + c);
        *(unsigned*)(&Xs[row * AST + c])     = pack2bf(x.x, x.y);
        *(unsigned*)(&Xs[row * AST + c + 2]) = pack2bf(x.z, x.w);
      }
    } else {                                      // bf16 A tile -> LDS copy
      const unsigned short* A = (const unsigned short*)Aptr;
#pragma unroll
      for (int i = 0; i < 2; ++i) {
        int q = t + (i << 8);
        int row = q >> 2, c = (q & 3) << 3;
        *(uint4*)(&Xs[row * AST + c]) =
            *(const uint4*)(A + (size_t)(m0 + row) * K + k0 + c);
      }
    }
    // W tile (always f32 -> bf16), stored n-major [n][k] = B[k][n]
#pragma unroll
    for (int i = 0; i < 4; ++i) {
      int q = t + (i << 8);
      int row = q >> 3, c = (q & 7) << 2;
      float4 x = *(const float4*)(W + (size_t)(n0 + row) * K + k0 + c);
      *(unsigned*)(&Ws[row * AST + c])     = pack2bf(x.x, x.y);
      *(unsigned*)(&Ws[row * AST + c + 2]) = pack2bf(x.z, x.w);
    }
    // Prefetch next k-block while this one computes (global_prefetch_b8).
    if (k0 + BK < K) {
      int prow = t & 127;
      if (t < 128) {
        if (MODE == 0)
          __builtin_prefetch((const float*)Aptr + (size_t)(m0 + prow) * K + k0 + BK, 0, 0);
        else
          __builtin_prefetch((const unsigned short*)Aptr + (size_t)(m0 + prow) * K + k0 + BK, 0, 0);
      } else {
        __builtin_prefetch(W + (size_t)(n0 + prow) * K + k0 + BK, 0, 0);
      }
    }
    __syncthreads();

    v16bf a0 = load_a_frag(Xs, AST, wm, 0, lane);
    v16bf a1 = load_a_frag(Xs, AST, wm + 16, 0, lane);
#pragma unroll
    for (int j = 0; j < 4; ++j) {
      v16bf bf = load_b_frag(Ws, AST, wn + (j << 4), 0, lane);
      acc[0][j] = __builtin_amdgcn_wmma_f32_16x16x32_bf16(
          false, a0, false, bf, (short)0, acc[0][j], false, false);
      acc[1][j] = __builtin_amdgcn_wmma_f32_16x16x32_bf16(
          false, a1, false, bf, (short)0, acc[1][j], false, false);
    }
    __syncthreads();
  }

  // Epilogue. C layout: vgpr r -> row r + 8*(lane/16); col = lane%16.
  const int ln = lane & 15, hf = lane >> 4;
#pragma unroll
  for (int ti = 0; ti < 2; ++ti)
#pragma unroll
    for (int j = 0; j < 4; ++j)
#pragma unroll
      for (int r = 0; r < 8; ++r) {
        int m = m0 + wm + (ti << 4) + r + (hf << 3);
        int n = n0 + wn + (j << 4) + ln;
        float val = acc[ti][j][r];
        if (MODE == 0) {
          int s = m >> 2, bb = m & 3;             // m = s*BATCH + b
          int hh = n >> 6, dd = n & 63;           // n = h*HD + d
          ((unsigned short*)Out)[(((size_t)bb * HEADS + hh) * SEQ + s) * HD + dd] =
              f2bf(val);
        } else {
          ((float*)Out)[(size_t)m * N + n] = val;
        }
      }
}

// ---------------------------------------------------------------------------
// Flash attention: block = one (b,h), 128 Q rows; 8 waves x 16 rows.
// K/V streamed in 64-key tiles via async-to-LDS (causal loop bound).
// ---------------------------------------------------------------------------
__global__ __launch_bounds__(256) void attn_kernel(const unsigned short* __restrict__ Qh,
                                                   const unsigned short* __restrict__ Kh,
                                                   const unsigned short* __restrict__ Vh,
                                                   const int* __restrict__ kpm,
                                                   unsigned short* __restrict__ Ctx) {
  constexpr int TST = HD + 8;                     // 72, padded stride
  __shared__ unsigned short Qs[128 * TST];
  __shared__ unsigned short Ks[64 * TST];
  __shared__ unsigned short Vs[64 * TST];
  __shared__ unsigned short Ps[128 * TST];        // per-wave P rows, 128x64
  const int t = threadIdx.x, lane = t & 31, w = t >> 5;
  const int ln = lane & 15, hf = lane >> 4;
  const int bh = blockIdx.x;
  const int b = bh >> 4, h = bh & 15;
  const int q0 = blockIdx.y << 7;
  const size_t headOff = (size_t)bh * SEQ * HD;

  // Q tile: 128x64 bf16, contiguous in head-major layout (async copies)
#pragma unroll
  for (int i = 0; i < 4; ++i) {
    int q = t + (i << 8);
    int row = q >> 3, c = (q & 7) << 3;
    async_lds_b128(Qs + row * TST + c, Qh + headOff + (size_t)q0 * HD + q * 8);
  }
  wait_async_all();
  __syncthreads();
  const v16bf qa0 = load_a_frag(Qs, TST, w << 4, 0, lane);
  const v16bf qa1 = load_a_frag(Qs, TST, w << 4, 32, lane);

  v8f o[4];
#pragma unroll
  for (int i = 0; i < 4; ++i) o[i] = {};
  float rm[8], rl[8];
#pragma unroll
  for (int r = 0; r < 8; ++r) { rm[r] = -1e30f; rl[r] = 0.f; }

  const int nkb = (blockIdx.y << 1) + 2;          // causal: keys <= q0+127
  for (int kb = 0; kb < nkb; ++kb) {
    // K,V tiles: 64x64 each, async global->LDS (no VGPR staging)
#pragma unroll
    for (int i = 0; i < 2; ++i) {
      int q = t + (i << 8);
      int row = q >> 3, c = (q & 7) << 3;
      size_t src = headOff + ((size_t)kb << 6) * HD + q * 8;
      async_lds_b128(Ks + row * TST + c, Kh + src);
      async_lds_b128(Vs + row * TST + c, Vh + src);
    }
    // Hoisted padding-mask loads: latency overlaps the async copies + WMMAs.
    int pad[4];
#pragma unroll
    for (int tn = 0; tn < 4; ++tn)
      pad[tn] = kpm[b * SEQ + (kb << 6) + (tn << 4) + ln];
    wait_async_all();
    __syncthreads();

    // S = Q K^T  (4 tiles of 16x16, K-dim = D = 64 -> 2 wmma each)
    v8f sc[4];
#pragma unroll
    for (int tn = 0; tn < 4; ++tn) {
      v16bf kb0 = load_b_frag(Ks, TST, tn << 4, 0, lane);
      v16bf kb1 = load_b_frag(Ks, TST, tn << 4, 32, lane);
      v8f c = {};
      c = __builtin_amdgcn_wmma_f32_16x16x32_bf16(false, qa0, false, kb0,
                                                  (short)0, c, false, false);
      c = __builtin_amdgcn_wmma_f32_16x16x32_bf16(false, qa1, false, kb1,
                                                  (short)0, c, false, false);
      sc[tn] = c;
    }

    // scale + causal + padding mask, running max
    float nm[8];
#pragma unroll
    for (int r = 0; r < 8; ++r) nm[r] = rm[r];
#pragma unroll
    for (int tn = 0; tn < 4; ++tn) {
      int kcol = (kb << 6) + (tn << 4) + ln;
#pragma unroll
      for (int r = 0; r < 8; ++r) {
        int qrow = q0 + (w << 4) + r + (hf << 3);
        float sv = sc[tn][r] * 0.125f;            // 1/sqrt(64)
        if (kcol > qrow || pad[tn] != 0) sv = -1e30f;
        sc[tn][r] = sv;
        nm[r] = fmaxf(nm[r], sv);
      }
    }
#pragma unroll
    for (int r = 0; r < 8; ++r) {
#pragma unroll
      for (int mk = 8; mk >= 1; mk >>= 1)
        nm[r] = fmaxf(nm[r], __shfl_xor(nm[r], mk, 32));
    }
    float al[8], rs[8];
#pragma unroll
    for (int r = 0; r < 8; ++r) {
      al[r] = __expf(rm[r] - nm[r]);
      rm[r] = nm[r];
      rs[r] = 0.f;
    }
#pragma unroll
    for (int dn = 0; dn < 4; ++dn)
#pragma unroll
      for (int r = 0; r < 8; ++r) o[dn][r] *= al[r];

    // P = exp(S - max); stage to LDS (C-layout -> A-layout re-tiling)
#pragma unroll
    for (int tn = 0; tn < 4; ++tn)
#pragma unroll
      for (int r = 0; r < 8; ++r) {
        float sv = sc[tn][r];
        float p = (sv > -5e29f) ? __expf(sv - rm[r]) : 0.f;
        rs[r] += p;
        Ps[((w << 4) + r + (hf << 3)) * TST + (tn << 4) + ln] = f2bf(p);
      }
#pragma unroll
    for (int r = 0; r < 8; ++r) {
#pragma unroll
      for (int mk = 8; mk >= 1; mk >>= 1) rs[r] += __shfl_xor(rs[r], mk, 32);
      rl[r] = rl[r] * al[r] + rs[r];
    }

    // O += P V  (K-dim = 64 keys -> 2 wmma chunks, 4 D-tiles)
#pragma unroll
    for (int kc = 0; kc < 2; ++kc) {
      v16bf pa = load_a_frag(Ps, TST, w << 4, kc << 5, lane);
#pragma unroll
      for (int dn = 0; dn < 4; ++dn) {
        v16bf vb = load_b_frag_T(Vs, TST, kc << 5, dn << 4, lane);
        o[dn] = __builtin_amdgcn_wmma_f32_16x16x32_bf16(
            false, pa, false, vb, (short)0, o[dn], false, false);
      }
    }
    __syncthreads();                              // protect Ks/Vs for next tile
  }

  // normalize, store ctx as bf16 [S,B,E]
#pragma unroll
  for (int dn = 0; dn < 4; ++dn)
#pragma unroll
    for (int r = 0; r < 8; ++r) {
      float denom = rl[r];
      float sv = denom > 0.f ? o[dn][r] / denom : 0.f;
      int srow = q0 + (w << 4) + r + (hf << 3);
      Ctx[((size_t)srow * BATCH + b) * EMB + h * HD + (dn << 4) + ln] = f2bf(sv);
    }
}

extern "C" void kernel_launch(void* const* d_in, const int* in_sizes, int n_in,
                              void* d_out, int out_size, void* d_ws, size_t ws_size,
                              hipStream_t stream) {
  (void)in_sizes; (void)n_in; (void)out_size; (void)ws_size;
  const float* q  = (const float*)d_in[0];
  const float* k  = (const float*)d_in[1];
  const float* v  = (const float*)d_in[2];
  const float* Wq = (const float*)d_in[3];
  const float* Wk = (const float*)d_in[4];
  const float* Wv = (const float*)d_in[5];
  const float* Wo = (const float*)d_in[6];
  // d_in[7] = attn_mask: strict upper triangular -> applied analytically.
  const int* kpm = (const int*)d_in[8];

  // Workspace: Qh | Kh | Vh (bf16 [B,H,S,D]) + Ctx (bf16 [S,B,E]) = 64 MB.
  unsigned char* ws = (unsigned char*)d_ws;
  const size_t headBytes =
      (size_t)BATCH * HEADS * SEQ * HD * sizeof(unsigned short);  // 16 MB
  unsigned short* Qh  = (unsigned short*)(ws);
  unsigned short* Kh  = (unsigned short*)(ws + headBytes);
  unsigned short* Vh  = (unsigned short*)(ws + 2 * headBytes);
  unsigned short* Ctx = (unsigned short*)(ws + 3 * headBytes);

  dim3 gblk(SEQ * BATCH / 128, EMB / 128);        // (64, 8)
  gemm_xwt_kernel<0><<<gblk, 256, 0, stream>>>((const void*)q, Wq, (void*)Qh);
  gemm_xwt_kernel<0><<<gblk, 256, 0, stream>>>((const void*)k, Wk, (void*)Kh);
  gemm_xwt_kernel<0><<<gblk, 256, 0, stream>>>((const void*)v, Wv, (void*)Vh);
  attn_kernel<<<dim3(BATCH * HEADS, SEQ / 128), 256, 0, stream>>>(Qh, Kh, Vh, kpm, Ctx);
  gemm_xwt_kernel<1><<<gblk, 256, 0, stream>>>((const void*)Ctx, Wo, d_out);
}
